// EnhancedMultiHeadAttention_76106820485170
// MI455X (gfx1250) — matrix operand order repaired and checked by reading, not compiled, not run on hardware
//
#include <hip/hip_runtime.h>
#include <hip/hip_bf16.h>

typedef __attribute__((ext_vector_type(16))) __bf16 v16bf;
typedef __attribute__((ext_vector_type(8)))  float  v8f;

#define NH    16
#define HD    64
#define SEQ   2048
#define HID   1024
#define BATCH 2

__device__ inline v8f wmma_bf16(v16bf a, v16bf b, v8f c) {
  return __builtin_amdgcn_wmma_f32_16x16x32_bf16(false, a, false, b, (short)0, c,
                                                 false, false);
}

// A-matrix fragment: 16(M) x 32(K), source row-major with stride lda (bf16).
// Layout: lanes 0-15 row=lane, K {0..15}; lanes 16-31 row=lane-16, K {16..31}
__device__ inline v16bf load_a_bf16(const __bf16* __restrict__ A, int lda) {
  int lane = threadIdx.x & 31;
  int half = lane >> 4;
  int row  = lane & 15;
  const __bf16* p = A + (size_t)row * lda + half * 16;
  v16bf r;
#pragma unroll
  for (int i = 0; i < 8; ++i) r[i] = p[i];
#pragma unroll
  for (int i = 0; i < 8; ++i) r[8 + i] = p[8 + i];
  return r;
}

// Same A fragment but source is fp32 (converted to bf16 on load); works for LDS ptrs.
__device__ inline v16bf load_a_f32(const float* A, int lda) {
  int lane = threadIdx.x & 31;
  int half = lane >> 4;
  int row  = lane & 15;
  const float* p = A + (size_t)row * lda + half * 16;
  v16bf r;
#pragma unroll
  for (int i = 0; i < 8; ++i) r[i] = (__bf16)p[i];
#pragma unroll
  for (int i = 0; i < 8; ++i) r[8 + i] = (__bf16)p[8 + i];
  return r;
}

// B-matrix fragment: 32(K) x 16(N), element (k,n) stored at B[n*ldb + k]
// (source row-major [N,K]: weight W[n,k] or K-matrix K[key,d]) -> contiguous loads.
__device__ inline v16bf load_b_rowmajN(const __bf16* __restrict__ B, int ldb) {
  int lane = threadIdx.x & 31;
  int half = lane >> 4;
  int n    = lane & 15;
  const __bf16* p = B + (size_t)n * ldb + half * 16;
  v16bf r;
#pragma unroll
  for (int i = 0; i < 16; ++i) r[i] = p[i];
  return r;
}

// B-matrix fragment: 32(K) x 16(N), element (k,n) stored at B[k*ldb + n] (e.g. V[key,d])
__device__ inline v16bf load_b_rowmajK(const __bf16* __restrict__ B, int ldb) {
  int lane = threadIdx.x & 31;
  int half = lane >> 4;
  int n    = lane & 15;
  const __bf16* p = B + n + (size_t)(half * 16) * ldb;
  v16bf r;
#pragma unroll
  for (int i = 0; i < 16; ++i) r[i] = p[(size_t)i * ldb];
  return r;
}

__global__ void cvt_f32_bf16(const float* __restrict__ in, __bf16* __restrict__ out,
                             int n) {
  int i = blockIdx.x * blockDim.x + threadIdx.x;
  if (i < n) out[i] = (__bf16)in[i];
}

// C[M,N] = A[M,1024](bf16) @ W[1024,1024]^T(bf16) + bias.
// K = N = HID fixed at compile time so all six fragment streams share just TWO
// per-lane base address registers; tile offsets (16K/32K/48K elems = 32/64/96KB)
// fold into the 24-bit global_load immediate -> no address spills.
// Block = 256 threads (8 waves). Wave computes 32x64 (2 M x 4 N tiles),
// software-pipelined over K.
__global__ void gemm_bias(const __bf16* __restrict__ A, const __bf16* __restrict__ W,
                          const float* __restrict__ bias, __bf16* __restrict__ outb,
                          float* __restrict__ outf) {
  constexpr int K = HID;
  constexpr int N = HID;
  int wave = threadIdx.x >> 5;
  int m0 = blockIdx.x * 256 + wave * 32;
  int n0 = blockIdx.y * 64;
  int lane = threadIdx.x & 31;
  int half = lane >> 4;
  int ln   = lane & 15;   // A row within tile; B column within tile

  // Two lane-variant base pointers; everything else is an immediate offset.
  const __bf16* aBase = A + (size_t)(m0 + ln) * K + half * 16;
  const __bf16* bBase = W + (size_t)(n0 + ln) * K + half * 16;

  auto loadA = [&](int off) {            // off: element offset (k0 or 16*K + k0)
    const __bf16* p = aBase + off;
    v16bf r;
#pragma unroll
    for (int i = 0; i < 8; ++i) r[i] = p[i];
#pragma unroll
    for (int i = 0; i < 8; ++i) r[8 + i] = p[8 + i];
    return r;
  };
  auto loadB = [&](int off) {
    const __bf16* p = bBase + off;
    v16bf r;
#pragma unroll
    for (int i = 0; i < 16; ++i) r[i] = p[i];
    return r;
  };

  v8f acc[2][4] = {};
  // prologue: fragments for k0 = 0
  v16bf a0 = loadA(0);
  v16bf a1 = loadA(16 * K);
  v16bf b0 = loadB(0);
  v16bf b1 = loadB(16 * K);
  v16bf b2 = loadB(32 * K);
  v16bf b3 = loadB(48 * K);

#pragma unroll 4
  for (int k0 = 32; k0 < K; k0 += 32) {
    // issue next-iteration loads first so they overlap this iteration's WMMAs
    v16bf a0n = loadA(k0);
    v16bf a1n = loadA(16 * K + k0);
    v16bf b0n = loadB(k0);
    v16bf b1n = loadB(16 * K + k0);
    v16bf b2n = loadB(32 * K + k0);
    v16bf b3n = loadB(48 * K + k0);

    acc[0][0] = wmma_bf16(a0, b0, acc[0][0]);
    acc[0][1] = wmma_bf16(a0, b1, acc[0][1]);
    acc[0][2] = wmma_bf16(a0, b2, acc[0][2]);
    acc[0][3] = wmma_bf16(a0, b3, acc[0][3]);
    acc[1][0] = wmma_bf16(a1, b0, acc[1][0]);
    acc[1][1] = wmma_bf16(a1, b1, acc[1][1]);
    acc[1][2] = wmma_bf16(a1, b2, acc[1][2]);
    acc[1][3] = wmma_bf16(a1, b3, acc[1][3]);

    a0 = a0n; a1 = a1n; b0 = b0n; b1 = b1n; b2 = b2n; b3 = b3n;
  }
  // epilogue for final K chunk
  acc[0][0] = wmma_bf16(a0, b0, acc[0][0]);
  acc[0][1] = wmma_bf16(a0, b1, acc[0][1]);
  acc[0][2] = wmma_bf16(a0, b2, acc[0][2]);
  acc[0][3] = wmma_bf16(a0, b3, acc[0][3]);
  acc[1][0] = wmma_bf16(a1, b0, acc[1][0]);
  acc[1][1] = wmma_bf16(a1, b1, acc[1][1]);
  acc[1][2] = wmma_bf16(a1, b2, acc[1][2]);
  acc[1][3] = wmma_bf16(a1, b3, acc[1][3]);

#pragma unroll
  for (int mt = 0; mt < 2; ++mt) {
#pragma unroll
    for (int t = 0; t < 4; ++t) {
#pragma unroll
      for (int j = 0; j < 8; ++j) {
        int m  = m0 + mt * 16 + half * 8 + j;
        int nn = n0 + t * 16 + ln;
        float v = acc[mt][t][j] + bias[nn];
        size_t idx = (size_t)m * N + nn;
        if (outf) outf[idx] = v;
        if (outb) outb[idx] = (__bf16)v;
      }
    }
  }
}

// One WG per (b, h, 16-query tile). 256 threads = 8 waves.
// Phase 1: scores QK^T into LDS (fp32, 16 x 2048 = 128KB).
// Phase 2: masked/scaled exact softmax; normalized probs -> global attn output.
// Phase 3: context = P @ V via WMMA; P (unnormalized exp) read from LDS, per-row
//          1/sum folded into the accumulator -> no HBM re-read of probs.
__global__ void attn_kernel(const __bf16* __restrict__ Q, const __bf16* __restrict__ Kk,
                            const __bf16* __restrict__ V, const int* __restrict__ mask,
                            float* __restrict__ attn, __bf16* __restrict__ ctx_out) {
  const int S = SEQ, H = HID;
  int qt = blockIdx.x, h = blockIdx.y, b = blockIdx.z;
  int q0 = qt * 16;

  __shared__ float sc[16 * SEQ];   // 128 KB score block (CDNA5: 320KB LDS/WG)
  __shared__ float red[16 * 16];
  __shared__ float ctx[16 * HD];

  const __bf16* Qbase = Q  + ((size_t)(b * S + q0)) * H + h * HD;
  const __bf16* Kbase = Kk + (size_t)b * S * H + h * HD;
  const __bf16* Vbase = V  + (size_t)b * S * H + h * HD;

  int wave = threadIdx.x >> 5;
  int lane = threadIdx.x & 31;
  int half = lane >> 4;
  int ln   = lane & 15;

  // Q tile (16 x 64) as two 16x32 A-fragments, reused across all key tiles
  v16bf qa0 = load_a_bf16(Qbase, H);
  v16bf qa1 = load_a_bf16(Qbase + 32, H);

  for (int kt = wave; kt < S / 16; kt += 8) {
    int key0 = kt * 16;
    v8f accS = {};
    v16bf kb0 = load_b_rowmajN(Kbase + (size_t)key0 * H, H);       // d = 0..31
    accS = wmma_bf16(qa0, kb0, accS);
    v16bf kb1 = load_b_rowmajN(Kbase + (size_t)key0 * H + 32, H);  // d = 32..63
    accS = wmma_bf16(qa1, kb1, accS);
#pragma unroll
    for (int j = 0; j < 8; ++j)
      sc[(half * 8 + j) * SEQ + key0 + ln] = accS[j];
  }
  for (int i = threadIdx.x; i < 16 * HD; i += 256) ctx[i] = 0.0f;
  __syncthreads();

  // ---- softmax: 16 threads per row, rows 0..15 ----
  int row = threadIdx.x >> 4;
  int sub = threadIdx.x & 15;
  const float scale = 0.125f;  // hd^-0.5, hd = 64
  float mx = -3.0e38f;
  for (int c = sub; c < S; c += 16) {
    float v = sc[row * SEQ + c] * scale;
    if (mask[b * S + c] == 0) v = -10000.0f;
    sc[row * SEQ + c] = v;
    mx = fmaxf(mx, v);
  }
  red[row * 16 + sub] = mx;
  __syncthreads();
  if (sub == 0) {
    float m = red[row * 16];
#pragma unroll
    for (int i = 1; i < 16; ++i) m = fmaxf(m, red[row * 16 + i]);
    red[row * 16] = m;
  }
  __syncthreads();
  mx = red[row * 16];
  float sum = 0.0f;
  for (int c = sub; c < S; c += 16) {
    float e = __expf(sc[row * SEQ + c] - mx);
    sc[row * SEQ + c] = e;        // sc now holds unnormalized exp values
    sum += e;
  }
  __syncthreads();
  red[row * 16 + sub] = sum;
  __syncthreads();
  if (sub == 0) {
    float s = 0.0f;
#pragma unroll
    for (int i = 0; i < 16; ++i) s += red[row * 16 + i];
    red[row * 16] = s;            // red[m*16] = row sum, stable from here on
  }
  __syncthreads();
  float inv = 1.0f / red[row * 16];
  float* arow = attn + ((size_t)(b * NH + h) * S + q0 + row) * S;
  for (int c = sub; c < S; c += 16)
    arow[c] = sc[row * SEQ + c] * inv;   // normalized attn weights -> output

  // ---- PV: 8 waves = 4 n-tiles x 2 K-halves, K = 2048, P from LDS ----
  int nt = wave & 3;
  int kh = wave >> 2;
  v8f accC = {};
  for (int ks = 0; ks < 32; ++ks) {
    int k0 = (kh * 32 + ks) * 32;
    v16bf pa = load_a_f32(&sc[k0], SEQ);                 // ds loads, bf16 convert
    v16bf vb = load_b_rowmajK(Vbase + (size_t)k0 * H, H);
    accC = wmma_bf16(pa, vb, accC);
  }
#pragma unroll
  for (int j = 0; j < 8; ++j) {
    int m = half * 8 + j;
    float invm = 1.0f / red[m * 16];     // fold softmax normalization in here
    atomicAdd(&ctx[m * HD + nt * 16 + ln], accC[j] * invm);
  }
  __syncthreads();

  for (int i = threadIdx.x; i < 16 * HD; i += 256) {
    int r = i / HD, d = i % HD;
    ctx_out[((size_t)(b * S + q0 + r)) * H + h * HD + d] = (__bf16)ctx[i];
  }
}

__global__ void gate_mul(const float* __restrict__ proj, const float* __restrict__ gat,
                         float* __restrict__ out, int n) {
  int i = blockIdx.x * blockDim.x + threadIdx.x;
  if (i < n) {
    float g = 1.0f / (1.0f + __expf(-gat[i]));
    out[i] = proj[i] * g;
  }
}

extern "C" void kernel_launch(void* const* d_in, const int* in_sizes, int n_in,
                              void* d_out, int out_size, void* d_ws, size_t ws_size,
                              hipStream_t stream) {
  const float* x    = (const float*)d_in[0];
  const int*   mask = (const int*)  d_in[1];
  const float* wq   = (const float*)d_in[2];
  const float* bq   = (const float*)d_in[3];
  const float* wk   = (const float*)d_in[4];
  const float* bk   = (const float*)d_in[5];
  const float* wv   = (const float*)d_in[6];
  const float* bv   = (const float*)d_in[7];
  const float* wo   = (const float*)d_in[8];
  const float* bo   = (const float*)d_in[9];
  const float* wg   = (const float*)d_in[10];
  const float* bg   = (const float*)d_in[11];

  const int B = BATCH, S = SEQ, H = HID;
  const size_t MS = (size_t)B * S;   // 4096 rows
  const size_t nX = MS * H;          // 4,194,304
  const size_t nW = (size_t)H * H;   // 1,048,576

  char* ws = (char*)d_ws;
  __bf16* xb   = (__bf16*)ws;  ws += nX * 2;
  __bf16* wqb  = (__bf16*)ws;  ws += nW * 2;
  __bf16* wkb  = (__bf16*)ws;  ws += nW * 2;
  __bf16* wvb  = (__bf16*)ws;  ws += nW * 2;
  __bf16* wob  = (__bf16*)ws;  ws += nW * 2;
  __bf16* wgb  = (__bf16*)ws;  ws += nW * 2;
  __bf16* Qb   = (__bf16*)ws;  ws += nX * 2;
  __bf16* Kb   = (__bf16*)ws;  ws += nX * 2;
  __bf16* Vb   = (__bf16*)ws;  ws += nX * 2;
  __bf16* Cb   = (__bf16*)ws;  ws += nX * 2;
  float*  proj = (float*)ws;   ws += nX * 4;
  float*  gat  = (float*)ws;   ws += nX * 4;
  // total workspace ~82 MB

  float* outp = (float*)d_out;
  float* attn = outp + nX;   // [B, NH, S, S] fp32, 537 MB (the roofline term)

  cvt_f32_bf16<<<(int)((nX + 255) / 256), 256, 0, stream>>>(x,  xb,  (int)nX);
  cvt_f32_bf16<<<(int)((nW + 255) / 256), 256, 0, stream>>>(wq, wqb, (int)nW);
  cvt_f32_bf16<<<(int)((nW + 255) / 256), 256, 0, stream>>>(wk, wkb, (int)nW);
  cvt_f32_bf16<<<(int)((nW + 255) / 256), 256, 0, stream>>>(wv, wvb, (int)nW);
  cvt_f32_bf16<<<(int)((nW + 255) / 256), 256, 0, stream>>>(wo, wob, (int)nW);
  cvt_f32_bf16<<<(int)((nW + 255) / 256), 256, 0, stream>>>(wg, wgb, (int)nW);

  dim3 gemmGrid((unsigned)(MS / 256), (unsigned)(H / 64));
  gemm_bias<<<gemmGrid, 256, 0, stream>>>(xb, wqb, bq, Qb, nullptr);
  gemm_bias<<<gemmGrid, 256, 0, stream>>>(xb, wkb, bk, Kb, nullptr);
  gemm_bias<<<gemmGrid, 256, 0, stream>>>(xb, wvb, bv, Vb, nullptr);

  attn_kernel<<<dim3(S / 16, NH, B), 256, 0, stream>>>(Qb, Kb, Vb, mask, attn, Cb);

  gemm_bias<<<gemmGrid, 256, 0, stream>>>(Cb, wob, bo, nullptr, proj);
  gemm_bias<<<gemmGrid, 256, 0, stream>>>(Cb, wgb, bg, nullptr, gat);

  gate_mul<<<(int)((nX + 255) / 256), 256, 0, stream>>>(proj, gat, outp, (int)nX);
}
